// LocalLatentsTransformer_36807869727012
// MI455X (gfx1250) — compile-verified, matrix operand order checked by reading, not run on hardware
//
#include <hip/hip_runtime.h>
#include <math.h>

// ---------------- problem constants (match reference) ----------------
#define B_   2
#define N_   768
#define D_   512
#define H_   8
#define P_   64
#define C_   256
#define L_   6
#define DH_  64
#define LAT_ 8
#define NSEQ_ 8
#define NSP_  8
#define NRD_  16
#define K_   40              // 2*NSEQ + NSP + NRD
#define M_   (B_ * N_)       // 1536 rows

typedef unsigned short u16;
typedef __attribute__((ext_vector_type(16))) __bf16 v16bf;
typedef __attribute__((ext_vector_type(8)))  float  v8f;
typedef __attribute__((ext_vector_type(4)))  unsigned int v4u;

union Frag { v16bf v; v4u u[2]; };

// ---------------- bf16 helpers (round-to-nearest-even) ----------------
__device__ inline u16 f2b(float f) {
  union { float f; unsigned u; } c; c.f = f;
  unsigned r = c.u + 0x7FFFu + ((c.u >> 16) & 1u);
  return (u16)(r >> 16);
}
__device__ inline float b2f(u16 h) {
  union { unsigned u; float f; } c; c.u = ((unsigned)h) << 16;
  return c.f;
}

// ---------------- wave32 reductions ----------------
__device__ inline float wred_sum(float v) {
  #pragma unroll
  for (int o = 16; o > 0; o >>= 1) v += __shfl_xor(v, o, 32);
  return v;
}
__device__ inline float wred_max(float v) {
  #pragma unroll
  for (int o = 16; o > 0; o >>= 1) v = fmaxf(v, __shfl_xor(v, o, 32));
  return v;
}

// fragment load: two contiguous 16-byte chunks (ISA 16-bit A/B VGPR layout)
__device__ inline void ldfrag(Frag& f, const u16* p) {
  f.u[0] = *(const v4u*)(p);
  f.u[1] = *(const v4u*)(p + 16);
}

// =====================================================================
// Weight transpose + f32->bf16 convert:  src [R, Cc] row-major f32
//                                     -> dst [Cc, R] row-major bf16
// =====================================================================
__global__ void transpose_b16_kernel(const float* __restrict__ src,
                                     u16* __restrict__ dst, int R, int Cc) {
  int id = blockIdx.x * 256 + threadIdx.x;
  if (id >= R * Cc) return;
  int r = id / Cc, c = id % Cc;
  dst[(size_t)c * R + r] = f2b(src[(size_t)r * Cc + c]);
}

__global__ void convert_b16_kernel(const float* __restrict__ src,
                                   u16* __restrict__ dst, int n) {
  int id = blockIdx.x * 256 + threadIdx.x;
  if (id < n) dst[id] = f2b(src[id]);
}

__global__ void copyf_kernel(const float* __restrict__ src,
                             float* __restrict__ dst, int n) {
  int id = blockIdx.x * 256 + threadIdx.x;
  if (id < n) dst[id] = src[id];
}

// =====================================================================
// Neighbor construction: seq +-8 (clipped), spatial KNN-8 on CA, 16 random
// one thread per (b,n)
// =====================================================================
__global__ void neighbors_kernel(const float* __restrict__ ca,
                                 const unsigned char* __restrict__ mask,
                                 const int* __restrict__ rand_idx,
                                 int* __restrict__ IDX, int* __restrict__ VLD) {
  int id = blockIdx.x * blockDim.x + threadIdx.x;
  if (id >= B_ * N_) return;
  int b = id / N_, n = id % N_;
  int mbn = mask[b * N_ + n] ? 1 : 0;
  int* ip = IDX + (size_t)id * K_;
  int* vp = VLD + (size_t)id * K_;
  int slot = 0;
  // sequential neighbors
  for (int t = 0; t < 2 * NSEQ_; t++) {
    int off = (t < NSEQ_) ? (t - NSEQ_) : (t - NSEQ_ + 1);
    int base = n + off;
    int cl = min(max(base, 0), N_ - 1);
    int inr = (base >= 0 && base < N_) ? 1 : 0;
    ip[slot] = cl;
    vp[slot] = (inr && mask[b * N_ + cl] && mbn) ? 1 : 0;
    slot++;
  }
  // spatial KNN (top-8 smallest d2, mask invalid + self)
  float bx = ca[((size_t)b * N_ + n) * 3 + 0];
  float by = ca[((size_t)b * N_ + n) * 3 + 1];
  float bz = ca[((size_t)b * N_ + n) * 3 + 2];
  float bd[NSP_]; int bi[NSP_];
  #pragma unroll
  for (int i = 0; i < NSP_; i++) { bd[i] = 3e38f; bi[i] = -1; }
  for (int j = 0; j < N_; j++) {
    if (j == n) continue;
    if (!mask[b * N_ + j]) continue;
    float dx = ca[((size_t)b * N_ + j) * 3 + 0] - bx;
    float dy = ca[((size_t)b * N_ + j) * 3 + 1] - by;
    float dz = ca[((size_t)b * N_ + j) * 3 + 2] - bz;
    float d2 = dx * dx + dy * dy + dz * dz;
    if (d2 < bd[NSP_ - 1]) {
      int pos = NSP_ - 1;
      while (pos > 0 && bd[pos - 1] > d2) {
        bd[pos] = bd[pos - 1]; bi[pos] = bi[pos - 1]; pos--;
      }
      bd[pos] = d2; bi[pos] = j;
    }
  }
  for (int i = 0; i < NSP_; i++) {
    ip[slot] = (bi[i] >= 0) ? bi[i] : 0;
    vp[slot] = (bi[i] >= 0 && mbn) ? 1 : 0;
    slot++;
  }
  // random neighbors
  for (int r = 0; r < NRD_; r++) {
    int j = rand_idx[(size_t)id * NRD_ + r];
    ip[slot] = j;
    vp[slot] = (mask[b * N_ + j] && mbn) ? 1 : 0;
    slot++;
  }
}

// =====================================================================
// bf16 WMMA GEMM v2:  C[M,Nc] = A[M,Kd](bf16,row-major) x Wt'[Kd,Nc]
// Wt stored transposed [Nc,Kd] bf16 so both fragment streams are two
// contiguous b128 loads per lane.
// Per-wave register blocking: 16 rows x 64 cols (4 accumulators), each A
// fragment feeds 4 WMMAs. Register double-buffering across the k-loop:
// loads for chunk k+32 / k+64 issue before the WMMAs on chunk k so the
// matrix pipe overlaps the fragment loads instead of waiting on them.
// Block = 8 waves -> 16 x 512 output tile.
// Epilogues: 0 = f32 store, 1 = bf16 store, 2 = X += acc*mask (residual),
//            3 = gelu(acc) -> bf16
// =====================================================================
template <int EP>
__global__ __launch_bounds__(256) void gemm_bf16_kernel(
    const u16* __restrict__ A, const u16* __restrict__ Wt,
    int M, int Kd, int Nc,
    float* __restrict__ outF, u16* __restrict__ outB,
    float* __restrict__ Xr, const unsigned char* __restrict__ maskR) {
  int wave = threadIdx.x >> 5;
  int lane = threadIdx.x & 31;
  int Ntile = blockIdx.x * 512 + wave * 64;   // 4 sub-tiles of 16 cols
  int Mtile = blockIdx.y * 16;
  int half = lane >> 4;            // lanes 16-31 hold K+8 per ISA layout
  int l16 = lane & 15;
  int kbase = half * 8;
  const u16* Ap = A + (size_t)(Mtile + l16) * Kd + kbase;
  const u16* Bp = Wt + (size_t)(Ntile + l16) * Kd + kbase;
  const size_t bstr = (size_t)16 * Kd;        // next 16-col sub-tile

  v8f acc[4];
  #pragma unroll
  for (int t = 0; t < 4; t++) acc[t] = (v8f){0.f,0.f,0.f,0.f,0.f,0.f,0.f,0.f};

  Frag a0, a1, b0[4], b1[4];
  // preload even chunk (k = 0)
  ldfrag(a0, Ap);
  #pragma unroll
  for (int t = 0; t < 4; t++) ldfrag(b0[t], Bp + t * bstr);

  for (int kk = 0; kk < Kd; kk += 64) {
    if (kk + 128 < Kd) {                       // -> global_prefetch_b8
      __builtin_prefetch(Ap + kk + 128, 0, 3);
      __builtin_prefetch(Bp + kk + 128, 0, 3);
    }
    // issue odd-chunk loads before consuming even chunk
    ldfrag(a1, Ap + kk + 32);
    #pragma unroll
    for (int t = 0; t < 4; t++) ldfrag(b1[t], Bp + t * bstr + kk + 32);
    #pragma unroll
    for (int t = 0; t < 4; t++)
      acc[t] = __builtin_amdgcn_wmma_f32_16x16x32_bf16(
          false, a0.v, false, b0[t].v, (short)0, acc[t], false, false);
    // issue next even-chunk loads before consuming odd chunk
    if (kk + 64 < Kd) {
      ldfrag(a0, Ap + kk + 64);
      #pragma unroll
      for (int t = 0; t < 4; t++) ldfrag(b0[t], Bp + t * bstr + kk + 64);
    }
    #pragma unroll
    for (int t = 0; t < 4; t++)
      acc[t] = __builtin_amdgcn_wmma_f32_16x16x32_bf16(
          false, a1.v, false, b1[t].v, (short)0, acc[t], false, false);
  }

  int rbase = Mtile + half * 8;                // ISA C/D layout
  #pragma unroll
  for (int t = 0; t < 4; t++) {
    int col = Ntile + t * 16 + l16;
    #pragma unroll
    for (int r = 0; r < 8; r++) {
      int rowi = rbase + r;
      float v = acc[t][r];
      size_t o = (size_t)rowi * Nc + col;
      if (EP == 0) {
        outF[o] = v;
      } else if (EP == 1) {
        outB[o] = f2b(v);
      } else if (EP == 2) {
        float m = maskR[rowi] ? 1.f : 0.f;
        Xr[o] += v * m;
      } else {
        float g = 0.7978845608f * (v + 0.044715f * v * v * v);
        outB[o] = f2b(0.5f * v * (1.f + tanhf(g)));
      }
    }
  }
}

// =====================================================================
// LayerNorm over D=512 + optional adaLN modulation, write bf16.
// one block (256 thr) per row; each thread owns 2 elements.
// =====================================================================
template <int MOD>
__global__ __launch_bounds__(256) void ln_kernel(
    const float* __restrict__ X, const float* __restrict__ SS,
    u16* __restrict__ HB) {
  int row = blockIdx.x;
  int t = threadIdx.x;
  const float* xr = X + (size_t)row * D_;
  float x0 = xr[t], x1 = xr[t + 256];
  float s = x0 + x1, s2 = x0 * x0 + x1 * x1;
  __shared__ float sa[8], sb[8];
  s = wred_sum(s); s2 = wred_sum(s2);
  if ((t & 31) == 0) { sa[t >> 5] = s; sb[t >> 5] = s2; }
  __syncthreads();
  float ts = 0.f, tq = 0.f;
  #pragma unroll
  for (int i = 0; i < 8; i++) { ts += sa[i]; tq += sb[i]; }
  float mean = ts * (1.f / D_);
  float var = tq * (1.f / D_) - mean * mean;
  float rs = rsqrtf(var + 1e-5f);
  float h0 = (x0 - mean) * rs;
  float h1 = (x1 - mean) * rs;
  if (MOD) {
    const float* ssr = SS + (size_t)row * (2 * D_);
    h0 = h0 * (1.f + ssr[t])       + ssr[D_ + t];
    h1 = h1 * (1.f + ssr[t + 256]) + ssr[D_ + t + 256];
  }
  HB[(size_t)row * D_ + t]       = f2b(h0);
  HB[(size_t)row * D_ + t + 256] = f2b(h1);
}

// =====================================================================
// Sparse neighbor attention: one block per token, one wave per head.
// Per neighbor: LN(k) over DH=64 + fused q.k and pair-bias dots via
// wave32 shfl reductions; register-resident 40-way softmax; o -> bf16.
// =====================================================================
__global__ __launch_bounds__(256) void attn_kernel(
    const float* __restrict__ Q, const float* __restrict__ Kb,
    const float* __restrict__ Vb, const float* __restrict__ pair_rep,
    const float* __restrict__ Wb_l, const int* __restrict__ IDX,
    const int* __restrict__ VLD, u16* __restrict__ OB) {
  int row = blockIdx.x;                // b*N + n
  int b = row / N_;
  int n = row % N_;
  int h = threadIdx.x >> 5;
  int lane = threadIdx.x & 31;
  int d0 = lane * 2;
  // q layernorm over DH
  float2 qv = *(const float2*)(Q + (size_t)row * D_ + h * DH_ + d0);
  float s = wred_sum(qv.x + qv.y);
  float s2 = wred_sum(qv.x * qv.x + qv.y * qv.y);
  float qm = s * (1.f / DH_);
  float qvar = s2 * (1.f / DH_) - qm * qm;
  float qrs = rsqrtf(qvar + 1e-5f);
  float qn0 = (qv.x - qm) * qrs, qn1 = (qv.y - qm) * qrs;
  // pair-bias weights for this head (loop invariant)
  float wb0 = Wb_l[(size_t)d0 * H_ + h];
  float wb1 = Wb_l[(size_t)(d0 + 1) * H_ + h];
  const int* ip = IDX + (size_t)row * K_;
  const int* vp = VLD + (size_t)row * K_;
  float sA = -3e38f, sB = -3e38f;      // lane-distributed scores: kk, kk+32
  for (int kk = 0; kk < K_; kk++) {
    int j = ip[kk];
    float2 kv = *(const float2*)(Kb + ((size_t)b * N_ + j) * D_ + h * DH_ + d0);
    float ks = wred_sum(kv.x + kv.y);
    float kq = wred_sum(kv.x * kv.x + kv.y * kv.y);
    float km = ks * (1.f / DH_);
    float kvar = kq * (1.f / DH_) - km * km;
    float krs = rsqrtf(kvar + 1e-5f);
    float k0 = (kv.x - km) * krs, k1 = (kv.y - km) * krs;
    const float* pg = pair_rep + (((size_t)b * N_ + n) * N_ + j) * P_ + d0;
    float pd = wred_sum(qn0 * k0 + qn1 * k1);
    float pb = wred_sum(pg[0] * wb0 + pg[1] * wb1);
    float score = pd * 0.125f + pb;    // scale = 1/sqrt(64)
    if (!vp[kk]) score = -1e9f;
    if (kk < 32) { if (lane == kk) sA = score; }
    else         { if (lane == kk - 32) sB = score; }
  }
  // softmax over K_=40 (lanes 0..31 own sA, lanes 0..7 also own sB)
  float loc = fmaxf(sA, (lane < K_ - 32) ? sB : -3e38f);
  float mx = wred_max(loc);
  float eA = __expf(sA - mx);
  float eB = (lane < K_ - 32) ? __expf(sB - mx) : 0.f;
  float tot = wred_sum(eA + eB);
  float inv = 1.f / tot;
  float aA = eA * inv, aB = eB * inv;
  float ox = 0.f, oy = 0.f;
  for (int kk = 0; kk < K_; kk++) {
    float a = (kk < 32) ? __shfl(aA, kk, 32) : __shfl(aB, kk - 32, 32);
    int j = ip[kk];
    float2 vv = *(const float2*)(Vb + ((size_t)b * N_ + j) * D_ + h * DH_ + d0);
    ox += a * vv.x; oy += a * vv.y;
  }
  size_t o = (size_t)row * D_ + h * DH_ + d0;
  OB[o] = f2b(ox); OB[o + 1] = f2b(oy);
}

// =====================================================================
// Final head: out[row, 0:3] = hN @ W_ca, out[row, 3:11] = hN @ W_lat
// 11 waves per block (one per output column), wave32 dot over D=512.
// =====================================================================
__global__ void head_kernel(const u16* __restrict__ HN,
                            const float* __restrict__ Wca,
                            const float* __restrict__ Wlat,
                            float* __restrict__ out) {
  int row = blockIdx.x;
  int w = threadIdx.x >> 5;        // 0..10
  int lane = threadIdx.x & 31;
  float acc = 0.f;
  for (int d = lane; d < D_; d += 32) {
    float wv = (w < 3) ? Wca[(size_t)d * 3 + w] : Wlat[(size_t)d * LAT_ + (w - 3)];
    acc += b2f(HN[(size_t)row * D_ + d]) * wv;
  }
  acc = wred_sum(acc);
  if (lane == 0) out[(size_t)row * (3 + LAT_) + w] = acc;
}

// =====================================================================
// host launcher
// =====================================================================
extern "C" void kernel_launch(void* const* d_in, const int* in_sizes, int n_in,
                              void* d_out, int out_size, void* d_ws, size_t ws_size,
                              hipStream_t stream) {
  (void)in_sizes; (void)n_in; (void)out_size; (void)ws_size;
  const float* seqs   = (const float*)d_in[0];
  const float* cond   = (const float*)d_in[1];
  const float* ca     = (const float*)d_in[2];
  const float* pair   = (const float*)d_in[3];
  const unsigned char* mask = (const unsigned char*)d_in[4];
  const int*   rand_idx = (const int*)d_in[5];
  const float* Wq = (const float*)d_in[6];
  const float* Wk = (const float*)d_in[7];
  const float* Wv = (const float*)d_in[8];
  const float* Wo = (const float*)d_in[9];
  const float* Wb = (const float*)d_in[10];
  const float* Wc = (const float*)d_in[11];
  const float* W1 = (const float*)d_in[12];
  const float* W2 = (const float*)d_in[13];
  const float* Wca = (const float*)d_in[14];
  const float* Wlat = (const float*)d_in[15];
  float* out = (float*)d_out;

  // ---- workspace layout ----
  size_t off = 0;
  char* base = (char*)d_ws;
  auto alloc = [&](size_t bytes) -> void* {
    void* p = base + off;
    off += (bytes + 255) & ~(size_t)255;
    return p;
  };
  u16* WqT = (u16*)alloc((size_t)L_ * D_ * D_ * 2);
  u16* WkT = (u16*)alloc((size_t)L_ * D_ * D_ * 2);
  u16* WvT = (u16*)alloc((size_t)L_ * D_ * D_ * 2);
  u16* WoT = (u16*)alloc((size_t)L_ * D_ * D_ * 2);
  u16* WcT = (u16*)alloc((size_t)L_ * 2 * D_ * C_ * 2);
  u16* W1T = (u16*)alloc((size_t)L_ * 4 * D_ * D_ * 2);
  u16* W2T = (u16*)alloc((size_t)L_ * D_ * 4 * D_ * 2);
  u16* condB = (u16*)alloc((size_t)M_ * C_ * 2);
  float* X  = (float*)alloc((size_t)M_ * D_ * 4);
  u16* HB   = (u16*)alloc((size_t)M_ * D_ * 2);
  float* SS = (float*)alloc((size_t)M_ * 2 * D_ * 4);
  float* Qf = (float*)alloc((size_t)M_ * D_ * 4);
  float* Kf = (float*)alloc((size_t)M_ * D_ * 4);
  float* Vf = (float*)alloc((size_t)M_ * D_ * 4);
  u16* OB   = (u16*)alloc((size_t)M_ * D_ * 2);
  u16* GB   = (u16*)alloc((size_t)M_ * 4 * D_ * 2);
  int* IDX  = (int*)alloc((size_t)M_ * K_ * 4);
  int* VLD  = (int*)alloc((size_t)M_ * K_ * 4);

  // ---- one-time prep: weight transpose/convert, x init, neighbors ----
  for (int l = 0; l < L_; l++) {
    int nDD = D_ * D_;
    transpose_b16_kernel<<<(nDD + 255) / 256, 256, 0, stream>>>(
        Wq + (size_t)l * nDD, WqT + (size_t)l * nDD, D_, D_);
    transpose_b16_kernel<<<(nDD + 255) / 256, 256, 0, stream>>>(
        Wk + (size_t)l * nDD, WkT + (size_t)l * nDD, D_, D_);
    transpose_b16_kernel<<<(nDD + 255) / 256, 256, 0, stream>>>(
        Wv + (size_t)l * nDD, WvT + (size_t)l * nDD, D_, D_);
    transpose_b16_kernel<<<(nDD + 255) / 256, 256, 0, stream>>>(
        Wo + (size_t)l * nDD, WoT + (size_t)l * nDD, D_, D_);
    int nC = C_ * 2 * D_;
    transpose_b16_kernel<<<(nC + 255) / 256, 256, 0, stream>>>(
        Wc + (size_t)l * nC, WcT + (size_t)l * nC, C_, 2 * D_);
    int n1 = D_ * 4 * D_;
    transpose_b16_kernel<<<(n1 + 255) / 256, 256, 0, stream>>>(
        W1 + (size_t)l * n1, W1T + (size_t)l * n1, D_, 4 * D_);
    transpose_b16_kernel<<<(n1 + 255) / 256, 256, 0, stream>>>(
        W2 + (size_t)l * n1, W2T + (size_t)l * n1, 4 * D_, D_);
  }
  convert_b16_kernel<<<(M_ * C_ + 255) / 256, 256, 0, stream>>>(cond, condB, M_ * C_);
  copyf_kernel<<<(M_ * D_ + 255) / 256, 256, 0, stream>>>(seqs, X, M_ * D_);
  neighbors_kernel<<<(M_ + 255) / 256, 256, 0, stream>>>(ca, mask, rand_idx, IDX, VLD);

  // ---- layer loop ----
  dim3 blk(256);
  dim3 gCond(2 * D_ / 512, M_ / 16);   // Nc=1024
  dim3 gD(D_ / 512 ? D_ / 512 : 1, M_ / 16);  // Nc=512 -> 1 block wide
  dim3 g4D(4 * D_ / 512, M_ / 16);     // Nc=2048
  for (int l = 0; l < L_; l++) {
    // adaLN conditioning:  SS = cond @ Wc[l]
    gemm_bf16_kernel<0><<<gCond, blk, 0, stream>>>(
        condB, WcT + (size_t)l * 2 * D_ * C_, M_, C_, 2 * D_, SS, nullptr, nullptr, nullptr);
    // h = ln(x)*(1+sc)+sh  -> bf16
    ln_kernel<1><<<M_, 256, 0, stream>>>(X, SS, HB);
    // Q/K/V projections
    gemm_bf16_kernel<0><<<gD, blk, 0, stream>>>(
        HB, WqT + (size_t)l * D_ * D_, M_, D_, D_, Qf, nullptr, nullptr, nullptr);
    gemm_bf16_kernel<0><<<gD, blk, 0, stream>>>(
        HB, WkT + (size_t)l * D_ * D_, M_, D_, D_, Kf, nullptr, nullptr, nullptr);
    gemm_bf16_kernel<0><<<gD, blk, 0, stream>>>(
        HB, WvT + (size_t)l * D_ * D_, M_, D_, D_, Vf, nullptr, nullptr, nullptr);
    // sparse neighbor attention -> o (bf16)
    attn_kernel<<<M_, 256, 0, stream>>>(
        Qf, Kf, Vf, pair, Wb + (size_t)l * P_ * H_, IDX, VLD, OB);
    // x += (o @ Wo) * mask
    gemm_bf16_kernel<2><<<gD, blk, 0, stream>>>(
        OB, WoT + (size_t)l * D_ * D_, M_, D_, D_, nullptr, nullptr, X, mask);
    // transition MLP
    ln_kernel<0><<<M_, 256, 0, stream>>>(X, nullptr, HB);
    gemm_bf16_kernel<3><<<g4D, blk, 0, stream>>>(
        HB, W1T + (size_t)l * D_ * 4 * D_, M_, D_, 4 * D_, nullptr, GB, nullptr, nullptr);
    gemm_bf16_kernel<2><<<gD, blk, 0, stream>>>(
        GB, W2T + (size_t)l * D_ * 4 * D_, M_, 4 * D_, D_, nullptr, nullptr, X, mask);
  }

  // ---- final head ----
  ln_kernel<0><<<M_, 256, 0, stream>>>(X, nullptr, HB);
  head_kernel<<<M_, 11 * 32, 0, stream>>>(HB, Wca, Wlat, out);
}
